// NANS_KoBeM_45535243272330
// MI455X (gfx1250) — matrix operand order repaired and checked by reading, not compiled, not run on hardware
//
#include <hip/hip_runtime.h>
#include <math.h>

typedef __bf16 bf16;
typedef __attribute__((ext_vector_type(16))) bf16 v16bf;
typedef __attribute__((ext_vector_type(8)))  bf16 v8bf;
typedef __attribute__((ext_vector_type(8)))  float v8f;

// ---------------- f32 -> bf16 conversion (optional zero pad) ----------------
__global__ void k_cvt_bf16(const float* __restrict__ in, bf16* __restrict__ out,
                           int n_in, int n_out)
{
    int i = blockIdx.x * blockDim.x + threadIdx.x;
    if (i >= n_out) return;
    out[i] = (i < n_in) ? (bf16)in[i] : (bf16)0.0f;
}

// ---------------- embedding gather: qe (bf16) and qa = [qe|ae] (bf16) -------
__global__ void k_gather(const int* __restrict__ q_data, const int* __restrict__ a_data,
                         const float* __restrict__ q_embed, const float* __restrict__ a_embed,
                         bf16* __restrict__ qe_bf, bf16* __restrict__ qa_bf, int bt)
{
    int idx = blockIdx.x * blockDim.x + threadIdx.x;
    if (idx >= bt * 128) return;
    int r = idx >> 7, e = idx & 127;
    int q = q_data[r], a = a_data[r];
    float qv = q_embed[(size_t)q * 128 + e];
    float av = a_embed[(size_t)a * 128 + e];
    qe_bf[idx] = (bf16)qv;
    qa_bf[(size_t)r * 256 + e]       = (bf16)qv;
    qa_bf[(size_t)r * 256 + 128 + e] = (bf16)av;
}

// ---------------- bf16 WMMA GEMM: Y = act(A @ W^T + bias) -------------------
// A: (M,K) row-major bf16.  W: (Npad,K) row-major bf16.  Y ld = Nvalid.
// Block: 256 thr = 8 waves arranged 2(M) x 4(N).
// Per-wave tile: 64 (M) x 16*TN (N)  ->  4 x TN WMMA tiles, big A/B reuse.
// Grid: (M/128, Npad/(64*TN)).
// ACT: 0=none 1=tanh 2=sigmoid.  OUT_BF: bf16 vs f32 out.  NT: nontemporal f32 store.
template <int ACT, int OUT_BF, int TN, int NT>
__global__ __launch_bounds__(256)
void k_gemm_bf16(const bf16* __restrict__ A, const bf16* __restrict__ W,
                 const float* __restrict__ bias, void* __restrict__ Y,
                 int K, int Nvalid)
{
    const int lane = threadIdx.x & 31;
    const int wave = threadIdx.x >> 5;
    const int m0 = blockIdx.x * 128 + (wave >> 2) * 64;        // 2 M-waves
    const int n0 = blockIdx.y * (64 * TN) + (wave & 3) * (16 * TN); // 4 N-waves
    const int r  = lane & 15;
    const int kh = lane >> 4;

    v8f acc[4][TN] = {};
    for (int k0 = 0; k0 < K; k0 += 32) {
        v16bf af[4], bfr[TN];
#pragma unroll
        for (int i = 0; i < 4; ++i) {
            // A frag (16-bit 16x32 layout): K = k0 + 8*kh + {0..7} and +16
            const bf16* pa = A + (size_t)(m0 + 16 * i + r) * K + k0 + 8 * kh;
            ((v8bf*)&af[i])[0] = *(const v8bf*)(pa);
            ((v8bf*)&af[i])[1] = *(const v8bf*)(pa + 16);
        }
#pragma unroll
        for (int j = 0; j < TN; ++j) {
            // B frag (16-bit 32x16 layout): lane n holds K = k0 + 8*kh + {0..7} and +16
            const bf16* pb = W + (size_t)(n0 + 16 * j + r) * K + k0 + 8 * kh;
            ((v8bf*)&bfr[j])[0] = *(const v8bf*)(pb);
            ((v8bf*)&bfr[j])[1] = *(const v8bf*)(pb + 16);
        }
#pragma unroll
        for (int i = 0; i < 4; ++i)
#pragma unroll
            for (int j = 0; j < TN; ++j)
                acc[i][j] = __builtin_amdgcn_wmma_f32_16x16x32_bf16(
                    false, af[i], false, bfr[j], (short)0, acc[i][j], false, false);
    }
#pragma unroll
    for (int j = 0; j < TN; ++j) {
        int n = n0 + 16 * j + r;
        bool ok = (n < Nvalid);
        float bv = (ok && bias) ? bias[n] : 0.0f;
#pragma unroll
        for (int i = 0; i < 4; ++i) {
            int mb = m0 + 16 * i + 8 * kh;   // C/D layout: vgpr e -> row mb+e
#pragma unroll
            for (int e = 0; e < 8; ++e) {
                float v = acc[i][j][e] + bv;
                if (ACT == 1) v = tanhf(v);
                if (ACT == 2) v = 1.0f / (1.0f + __expf(-v));
                if (ok) {
                    size_t o = (size_t)(mb + e) * Nvalid + n;
                    if (OUT_BF)      ((bf16*)Y)[o] = (bf16)v;
                    else if (NT)     __builtin_nontemporal_store(v, (float*)Y + o);
                    else             ((float*)Y)[o] = v;
                }
            }
        }
    }
}

// ---------------- softmax over last dim of (rows,64), in place --------------
__global__ void k_softmax64(float* __restrict__ x, int rows)
{
    int r = blockIdx.x * blockDim.x + threadIdx.x;
    if (r >= rows) return;
    float* p = x + (size_t)r * 64;
    float v[64];
    float mx = -3.0e38f;
#pragma unroll
    for (int i = 0; i < 64; ++i) { v[i] = p[i]; mx = fmaxf(mx, v[i]); }
    float s = 0.0f;
#pragma unroll
    for (int i = 0; i < 64; ++i) { v[i] = __expf(v[i] - mx); s += v[i]; }
    float inv = 1.0f / s;
#pragma unroll
    for (int i = 0; i < 64; ++i) p[i] = v[i] * inv;
}

// ---------------- sequential memory scan; M (64x128) in LDS -----------------
// One block per batch element, 128 threads (thread = value column v).
__global__ __launch_bounds__(128)
void k_scan(const float* __restrict__ corr,      // (B,T,64)
            const float* __restrict__ erase_sig, // (B,T,128)
            const float* __restrict__ add_sig,   // (B,T,128)
            const float* __restrict__ value_init,// (64,128)
            const bf16*  __restrict__ qe_bf,     // (B,T,128)
            bf16* __restrict__ behavior,         // (B,T-1,256) = [rc | qe_t]
            bf16* __restrict__ mastery,          // (B,T-1,256) = [rc_next | qe_{t+1}]
            int T)
{
    __shared__ float M[64 * 128];
    __shared__ float w[64];
    __shared__ float wn[64];
    const int b = blockIdx.x;
    const int v = threadIdx.x;      // 0..127
    for (int c = 0; c < 64; ++c) M[c * 128 + v] = value_init[c * 128 + v];
    const int Tm1 = T - 1;
    for (int t = 0; t < Tm1; ++t) {
        __syncthreads();
        if (v < 64) w[v] = corr[((size_t)b * T + t) * 64 + v];
        else        wn[v - 64] = corr[((size_t)b * T + t + 1) * 64 + (v - 64)];
        __syncthreads();
        const float e = erase_sig[((size_t)b * T + t) * 128 + v];
        const float a = add_sig  [((size_t)b * T + t) * 128 + v];
        float rc = 0.0f, rcn = 0.0f;
#pragma unroll 4
        for (int c = 0; c < 64; ++c) {
            float wc = w[c];
            float m = M[c * 128 + v];
            m = m * (1.0f - wc * e) + wc * a;
            M[c * 128 + v] = m;
            rc  += wc * m;
            rcn += wn[c] * m;
        }
        size_t o = ((size_t)b * Tm1 + t) * 256;
        behavior[o + v]       = (bf16)rc;
        behavior[o + 128 + v] = qe_bf[((size_t)b * T + t) * 128 + v];
        mastery[o + v]        = (bf16)rcn;
        mastery[o + 128 + v]  = qe_bf[((size_t)b * T + t + 1) * 128 + v];
    }
}

// ---------------- pred head: sigmoid(h . Wo + bo), wave per row -------------
__global__ void k_pred_head(const bf16* __restrict__ h, const float* __restrict__ Wo,
                            const float* __restrict__ bo, float* __restrict__ out, int rows)
{
    int row  = (blockIdx.x * blockDim.x + threadIdx.x) >> 5;
    int lane = threadIdx.x & 31;
    if (row >= rows) return;
    const bf16* p = h + (size_t)row * 256;
    float s = 0.0f;
#pragma unroll
    for (int i = 0; i < 8; ++i) s += (float)p[lane + 32 * i] * Wo[lane + 32 * i];
    for (int off = 16; off > 0; off >>= 1) s += __shfl_down(s, off, 32);
    if (lane == 0) out[row] = 1.0f / (1.0f + __expf(-(s + bo[0])));
}

// ===========================================================================
extern "C" void kernel_launch(void* const* d_in, const int* in_sizes, int n_in,
                              void* d_out, int out_size, void* d_ws, size_t ws_size,
                              hipStream_t stream)
{
    (void)in_sizes; (void)n_in; (void)out_size; (void)ws_size;
    const int B = 128, T = 200, BT = B * T;            // 25600
    const int Tm1 = T - 1, NR = B * Tm1;               // 199, 25472
    const int NQ1 = 5001, NPAD = 5120;                 // Wob rows, padded to 20*256

    const int*   q_data  = (const int*)  d_in[0];
    const int*   a_data  = (const int*)  d_in[1];
    const float* q_embed = (const float*)d_in[2];
    const float* a_embed = (const float*)d_in[3];
    const float* key_mx  = (const float*)d_in[4];
    const float* val_ini = (const float*)d_in[5];
    const float* Wk  = (const float*)d_in[6];  const float* bk  = (const float*)d_in[7];
    const float* We  = (const float*)d_in[8];  const float* be  = (const float*)d_in[9];
    const float* We2 = (const float*)d_in[10]; const float* be2 = (const float*)d_in[11];
    const float* Wa  = (const float*)d_in[12]; const float* ba  = (const float*)d_in[13];
    const float* Wa2 = (const float*)d_in[14]; const float* ba2 = (const float*)d_in[15];
    const float* Ws  = (const float*)d_in[16]; const float* bs  = (const float*)d_in[17];
    const float* Wo  = (const float*)d_in[18]; const float* bo  = (const float*)d_in[19];
    const float* Wsb = (const float*)d_in[20]; const float* bsb = (const float*)d_in[21];
    const float* Wob = (const float*)d_in[22]; const float* bob = (const float*)d_in[23];

    // ---- workspace carve-up (256B aligned) ----
    char* p = (char*)d_ws;
    auto alloc = [&](size_t bytes) -> void* {
        void* r = (void*)p; p += (bytes + 255) & ~(size_t)255; return r;
    };
    bf16* wk_bf   = (bf16*)alloc((size_t)128 * 128 * 2);
    bf16* key_bf  = (bf16*)alloc((size_t)64  * 128 * 2);
    bf16* we_bf   = (bf16*)alloc((size_t)128 * 256 * 2);
    bf16* we2_bf  = (bf16*)alloc((size_t)128 * 128 * 2);
    bf16* wa_bf   = (bf16*)alloc((size_t)128 * 256 * 2);
    bf16* wa2_bf  = (bf16*)alloc((size_t)128 * 128 * 2);
    bf16* ws_bf   = (bf16*)alloc((size_t)256 * 256 * 2);
    bf16* wsb_bf  = (bf16*)alloc((size_t)256 * 256 * 2);
    bf16* wob_bf  = (bf16*)alloc((size_t)NPAD * 256 * 2);
    bf16* qe_bf   = (bf16*)alloc((size_t)BT * 128 * 2);
    bf16* qa_bf   = (bf16*)alloc((size_t)BT * 256 * 2);
    float* corr   = (float*)alloc((size_t)BT * 64 * 4);
    bf16* hid_bf  = (bf16*)alloc((size_t)BT * 128 * 2);   // reused intermediate
    float* ersig  = (float*)alloc((size_t)BT * 128 * 4);
    float* adsig  = (float*)alloc((size_t)BT * 128 * 4);
    bf16* beh_bf  = (bf16*)alloc((size_t)NR * 256 * 2);
    bf16* mas_bf  = (bf16*)alloc((size_t)NR * 256 * 2);
    bf16* h_bf    = (bf16*)alloc((size_t)NR * 256 * 2);

    float* pred          = (float*)d_out;          // (B, T-1)
    float* pred_material = (float*)d_out + NR;     // (B, T-1, 5001)

    auto cvt = [&](const float* src, bf16* dst, int n_in_, int n_out_) {
        k_cvt_bf16<<<(n_out_ + 255) / 256, 256, 0, stream>>>(src, dst, n_in_, n_out_);
    };
    cvt(Wk,     wk_bf,  128 * 128, 128 * 128);
    cvt(key_mx, key_bf, 64 * 128,  64 * 128);
    cvt(We,     we_bf,  128 * 256, 128 * 256);
    cvt(We2,    we2_bf, 128 * 128, 128 * 128);
    cvt(Wa,     wa_bf,  128 * 256, 128 * 256);
    cvt(Wa2,    wa2_bf, 128 * 128, 128 * 128);
    cvt(Ws,     ws_bf,  256 * 256, 256 * 256);
    cvt(Wsb,    wsb_bf, 256 * 256, 256 * 256);
    cvt(Wob,    wob_bf, NQ1 * 256, NPAD * 256);    // zero-padded tail

    k_gather<<<(BT * 128 + 255) / 256, 256, 0, stream>>>(
        q_data, a_data, q_embed, a_embed, qe_bf, qa_bf, BT);

    const int gBT = BT / 128;   // 200
    const int gNR = NR / 128;   // 199

    // read_keys = qe @ Wk^T + bk  -> bf16 (reuse hid_bf)     N=128 -> TN=2
    k_gemm_bf16<0, 1, 2, 0><<<dim3(gBT, 1), 256, 0, stream>>>(qe_bf, wk_bf, bk, hid_bf, 128, 128);
    // corr logits = read_keys @ key^T -> f32                 N=64  -> TN=1
    k_gemm_bf16<0, 0, 1, 0><<<dim3(gBT, 1), 256, 0, stream>>>(hid_bf, key_bf, nullptr, corr, 128, 64);
    k_softmax64<<<(BT + 255) / 256, 256, 0, stream>>>(corr, BT);

    // erase_sig = sigmoid((qa@We^T+be)@We2^T+be2)
    k_gemm_bf16<0, 1, 2, 0><<<dim3(gBT, 1), 256, 0, stream>>>(qa_bf, we_bf, be, hid_bf, 256, 128);
    k_gemm_bf16<2, 0, 2, 0><<<dim3(gBT, 1), 256, 0, stream>>>(hid_bf, we2_bf, be2, ersig, 128, 128);
    // add_sig = tanh((qa@Wa^T+ba)@Wa2^T+ba2)
    k_gemm_bf16<0, 1, 2, 0><<<dim3(gBT, 1), 256, 0, stream>>>(qa_bf, wa_bf, ba, hid_bf, 256, 128);
    k_gemm_bf16<1, 0, 2, 0><<<dim3(gBT, 1), 256, 0, stream>>>(hid_bf, wa2_bf, ba2, adsig, 128, 128);

    // sequential memory scan -> behavior / mastery (bf16, concat with qe tails)
    k_scan<<<B, 128, 0, stream>>>(corr, ersig, adsig, val_ini, qe_bf, beh_bf, mas_bf, T);

    // pred_material = sigmoid(tanh(behavior@Wsb^T+bsb) @ Wob^T + bob)
    k_gemm_bf16<1, 1, 4, 0><<<dim3(gNR, 1), 256, 0, stream>>>(beh_bf, wsb_bf, bsb, h_bf, 256, 256);
    k_gemm_bf16<2, 0, 4, 1><<<dim3(gNR, NPAD / 256), 256, 0, stream>>>(
        h_bf, wob_bf, bob, pred_material, 256, NQ1);

    // pred = sigmoid(tanh(mastery@Ws^T+bs) @ Wo^T + bo)
    k_gemm_bf16<1, 1, 4, 0><<<dim3(gNR, 1), 256, 0, stream>>>(mas_bf, ws_bf, bs, h_bf, 256, 256);
    k_pred_head<<<(NR * 32 + 255) / 256, 256, 0, stream>>>(h_bf, Wo, bo, pred, NR);
}
